// Seq2Seq_4707284157154
// MI455X (gfx1250) — compile-verified, matrix operand order checked by reading, not run on hardware
//
#include <hip/hip_runtime.h>

// ---------------------------------------------------------------------------
// Problem constants (match reference)
// ---------------------------------------------------------------------------
#define VOC 32000
#define EMB 512
#define HID 512
#define DDIM 512
#define BB 32
#define SS 64
#define TT 48

typedef __attribute__((ext_vector_type(16))) __bf16 bf16x16;
typedef __attribute__((ext_vector_type(8)))  __bf16 bf16x8;
typedef __attribute__((ext_vector_type(8)))  float  f32x8;

// ---------------------------------------------------------------------------
// Explicit global-address-space accessors. Generic kernarg pointers otherwise
// lower to flat_load/flat_store, which consume both LOADcnt and DScnt and
// couple the global stream to the LDS pipe. Casting through an integer into
// address_space(1) forces global_load_b128 / global_store_*.
// ---------------------------------------------------------------------------
typedef __attribute__((address_space(1))) bf16x8 g1_bf16x8;
typedef __attribute__((address_space(1))) float  g1_float;
typedef __attribute__((address_space(1))) __bf16 g1_bf16;

__device__ __forceinline__ bf16x8 gload8(const __bf16* p) {
  return *(const g1_bf16x8*)(unsigned long long)p;
}
__device__ __forceinline__ float gloadf(const float* p) {
  return *(const g1_float*)(unsigned long long)p;
}
__device__ __forceinline__ __bf16 gloadb(const __bf16* p) {
  return *(const g1_bf16*)(unsigned long long)p;
}
__device__ __forceinline__ void gstoref(float* p, float v) {
  *(g1_float*)(unsigned long long)p = v;
}
__device__ __forceinline__ void gstoreb(__bf16* p, __bf16 v) {
  *(g1_bf16*)(unsigned long long)p = v;
}

// ---------------------------------------------------------------------------
// WMMA operand loaders.
// 16-bit A layout (16x32, MxK): lane<16 holds K[0..7],K[16..23] of row M=lane;
// lane>=16 holds K[8..15],K[24..31] of row M=lane-16.
// W stored as [N,K] row-major loads identically for the B operand (B = W^T).
// base points at element [row][kk]; half = lane>>4.
// ---------------------------------------------------------------------------
__device__ __forceinline__ bf16x16 load_op16_lds(const __bf16* base, int half) {
  const bf16x8 lo = *reinterpret_cast<const bf16x8*>(base + half * 8);
  const bf16x8 hi = *reinterpret_cast<const bf16x8*>(base + 16 + half * 8);
  return __builtin_shufflevector(lo, hi, 0, 1, 2, 3, 4, 5, 6, 7,
                                 8, 9, 10, 11, 12, 13, 14, 15);
}

__device__ __forceinline__ bf16x16 load_op16_g(const __bf16* base, int half) {
  const bf16x8 lo = gload8(base + half * 8);
  const bf16x8 hi = gload8(base + 16 + half * 8);
  return __builtin_shufflevector(lo, hi, 0, 1, 2, 3, 4, 5, 6, 7,
                                 8, 9, 10, 11, 12, 13, 14, 15);
}

__device__ __forceinline__ f32x8 wmma_bf16(bf16x16 a, bf16x16 b, f32x8 c) {
  return __builtin_amdgcn_wmma_f32_16x16x32_bf16(false, a, false, b,
                                                 (short)0, c, false, false);
}

__device__ __forceinline__ float sigmoidf_(float x) {
  return 1.0f / (1.0f + __expf(-x));
}

// ---------------------------------------------------------------------------
// Generic GEMM: C[M,N] = act(A[M,K] @ W[N,K]^T + bias), fp32 out (+ optional
// bf16 shadow copy). 8 waves / block, one 16x16 tile per wave. All launches
// guarantee tilesN % 8 == 0, so every block's tiles share one M-tile: the 16
// A-rows are staged into dynamic LDS once; WMMA A streams from ds_load_b128,
// W streams from global (L2-resident for the hot weights). The K-loop is
// software-pipelined (next fragments loaded before current WMMA) so the
// backend keeps rotating operand buffers and overlaps loads with math.
// M,N multiples of 16; K multiple of 32. Dynamic LDS = 16*K*2 bytes.
// ---------------------------------------------------------------------------
template <int ACT, bool HAS_BIAS, bool WRITE_BF16>
__global__ __launch_bounds__(256) void gemm_wmma_kernel(
    const __bf16* __restrict__ A, int lda,
    const __bf16* __restrict__ W, int ldw,
    const float* __restrict__ bias,
    float* __restrict__ C, int ldc,
    __bf16* __restrict__ Cbf, int ldcbf,
    int M, int N, int K) {
  extern __shared__ __bf16 Ash[];  // [16, K]

  const int wave = threadIdx.x >> 5;
  const int lane = threadIdx.x & 31;
  const int tilesN = N >> 4;
  const int tilesM = M >> 4;
  const int tile0 = blockIdx.x * 8;
  const int tm = tile0 / tilesN;  // uniform across the block

  // ---- cooperative stage of the block's 16 A rows into LDS ----
  {
    const int kchunks = K >> 3;  // 8 bf16 = 16B per chunk
    for (int row = 0; row < 16; ++row) {
      const __bf16* srcp = A + (size_t)(tm * 16 + row) * lda;
      __bf16* dstp = Ash + row * K;
      for (int c = threadIdx.x; c < kchunks; c += 256)
        *reinterpret_cast<bf16x8*>(dstp + c * 8) = gload8(srcp + c * 8);
    }
  }
  __syncthreads();

  const int tile = tile0 + wave;
  if (tile >= tilesM * tilesN) return;
  const int tn = tile - tm * tilesN;
  const int half = lane >> 4;
  const int mr = lane & 15;

  const __bf16* Abase = Ash + mr * K;
  const __bf16* Wbase = W + (size_t)(tn * 16 + mr) * ldw;

  f32x8 acc;
  const float binit = HAS_BIAS ? gloadf(bias + tn * 16 + mr) : 0.0f;
#pragma unroll
  for (int r = 0; r < 8; ++r) acc[r] = binit;

  // software pipeline: prologue loads, steady state loads k+1 then mults k
  bf16x16 av = load_op16_lds(Abase, half);
  bf16x16 bv = load_op16_g(Wbase, half);
  for (int kk = 32; kk < K; kk += 32) {
    __builtin_prefetch(Wbase + kk + 96, 0, 1);  // 192B ahead on the W stream
    bf16x16 avn = load_op16_lds(Abase + kk, half);
    bf16x16 bvn = load_op16_g(Wbase + kk, half);
    acc = wmma_bf16(av, bv, acc);
    av = avn;
    bv = bvn;
  }
  acc = wmma_bf16(av, bv, acc);

  const int col = tn * 16 + mr;
#pragma unroll
  for (int r = 0; r < 8; ++r) {
    const int row = tm * 16 + r + half * 8;
    float v = acc[r];
    if (ACT == 1) v = tanhf(v);
    gstoref(C + (size_t)row * ldc + col, v);
    if (WRITE_BF16) gstoreb(Cbf + (size_t)row * ldcbf + col, (__bf16)v);
  }
}

// ---------------------------------------------------------------------------
// M=32-specialized GEMM: every wave computes BOTH M-tiles of one N column,
// reusing the B fragment in registers -> W reads from L2 are halved. All 32
// A rows staged in LDS (32*K*2 bytes dynamic). tilesN % 8 == 0 guaranteed.
// Used for the dominant pred GEMM (out_W, 47 replays) and all other M=32
// GEMMs (sterm, GiC, Gh, fc). K-loop software-pipelined as above.
// ---------------------------------------------------------------------------
template <int ACT, bool HAS_BIAS, bool WRITE_BF16>
__global__ __launch_bounds__(256) void gemm_wmma_m32_kernel(
    const __bf16* __restrict__ A, int lda,
    const __bf16* __restrict__ W, int ldw,
    const float* __restrict__ bias,
    float* __restrict__ C, int ldc,
    __bf16* __restrict__ Cbf, int ldcbf,
    int N, int K) {
  extern __shared__ __bf16 Ash[];  // [32, K]

  const int wave = threadIdx.x >> 5;
  const int lane = threadIdx.x & 31;

  {
    const int kchunks = K >> 3;
    for (int row = 0; row < 32; ++row) {
      const __bf16* srcp = A + (size_t)row * lda;
      __bf16* dstp = Ash + row * K;
      for (int c = threadIdx.x; c < kchunks; c += 256)
        *reinterpret_cast<bf16x8*>(dstp + c * 8) = gload8(srcp + c * 8);
    }
  }
  __syncthreads();

  const int tn = blockIdx.x * 8 + wave;  // tilesN % 8 == 0 -> always valid
  const int half = lane >> 4;
  const int mr = lane & 15;

  const __bf16* A0 = Ash + mr * K;         // rows 0..15
  const __bf16* A1 = Ash + (16 + mr) * K;  // rows 16..31
  const __bf16* Wbase = W + (size_t)(tn * 16 + mr) * ldw;

  f32x8 acc0, acc1;
  const float binit = HAS_BIAS ? gloadf(bias + tn * 16 + mr) : 0.0f;
#pragma unroll
  for (int r = 0; r < 8; ++r) {
    acc0[r] = binit;
    acc1[r] = binit;
  }

  bf16x16 bv = load_op16_g(Wbase, half);  // shared by both M-tiles
  bf16x16 a0 = load_op16_lds(A0, half);
  bf16x16 a1 = load_op16_lds(A1, half);
  for (int kk = 32; kk < K; kk += 32) {
    __builtin_prefetch(Wbase + kk + 96, 0, 1);
    bf16x16 bvn = load_op16_g(Wbase + kk, half);
    bf16x16 a0n = load_op16_lds(A0 + kk, half);
    bf16x16 a1n = load_op16_lds(A1 + kk, half);
    acc0 = wmma_bf16(a0, bv, acc0);
    acc1 = wmma_bf16(a1, bv, acc1);
    bv = bvn;
    a0 = a0n;
    a1 = a1n;
  }
  acc0 = wmma_bf16(a0, bv, acc0);
  acc1 = wmma_bf16(a1, bv, acc1);

  const int col = tn * 16 + mr;
#pragma unroll
  for (int r = 0; r < 8; ++r) {
    const int row0 = r + half * 8;
    const int row1 = 16 + r + half * 8;
    float v0 = acc0[r];
    float v1 = acc1[r];
    if (ACT == 1) {
      v0 = tanhf(v0);
      v1 = tanhf(v1);
    }
    gstoref(C + (size_t)row0 * ldc + col, v0);
    gstoref(C + (size_t)row1 * ldc + col, v1);
    if (WRITE_BF16) {
      gstoreb(Cbf + (size_t)row0 * ldcbf + col, (__bf16)v0);
      gstoreb(Cbf + (size_t)row1 * ldcbf + col, (__bf16)v1);
    }
  }
}

// ---------------------------------------------------------------------------
// fp32 -> bf16 conversion
// ---------------------------------------------------------------------------
__global__ void f2bf_kernel(const float* __restrict__ src,
                            __bf16* __restrict__ dst, int n) {
  int i = blockIdx.x * blockDim.x + threadIdx.x;
  if (i < n) gstoreb(dst + i, (__bf16)gloadf(src + i));
}

// ---------------------------------------------------------------------------
// Embedding gather -> bf16 rows.
// mode 0: dst row r = s*32+b  <- emb[src[b*64+s]]   (encoder, time-major)
// mode 1: dst row r = t*32+b  <- emb[trg[t*32+b]]   (decoder tokens)
// ---------------------------------------------------------------------------
__global__ void gather_embed_kernel(const int* __restrict__ toks,
                                    const float* __restrict__ emb,
                                    __bf16* __restrict__ dst, int mode) {
  const int r = blockIdx.x;
  int tokIdx;
  if (mode == 0) {
    const int s = r >> 5, b = r & 31;
    tokIdx = b * SS + s;
  } else {
    tokIdx = r;
  }
  const int tk = toks[tokIdx];
  const float* e = emb + (size_t)tk * EMB;
  for (int d = threadIdx.x; d < EMB; d += blockDim.x)
    gstoreb(dst + (size_t)r * EMB + d, (__bf16)gloadf(e + d));
}

// ---------------------------------------------------------------------------
// Bidirectional GRU encoder scan. grid.x = 2 (direction), 1024 threads.
// h kept in LDS as bf16 (WMMA A operand) + fp32 state in global ws.
// Per step: gh = h @ Whh^T + bhh via WMMA (192 tiles / 32 waves = 6 each),
// then fused gate update.
// The Whh pointer is laundered through an empty asm each time step so the
// compiler cannot hoist the (time-invariant) weight loads out of the scan
// and spill them to scratch; they stream from L2 instead.
// Gi layout: [S, B, 3H], row = s*B + b (bih already folded in by the GEMM).
// ---------------------------------------------------------------------------
__global__ __launch_bounds__(1024) void encoder_scan_kernel(
    const __bf16* __restrict__ WhhF, const __bf16* __restrict__ WhhB,
    const float* __restrict__ bhhF, const float* __restrict__ bhhB,
    const float* __restrict__ GiF, const float* __restrict__ GiB,
    float* __restrict__ enc_out,      // [B,S,2H]
    __bf16* __restrict__ enc_out_bf,  // [B,S,2H]
    __bf16* __restrict__ hcat,        // [B,2H] final hidden (bf16)
    float* __restrict__ h_ws,         // [2][B,H]
    float* __restrict__ gh_ws)        // [2][B,3H]
{
  __shared__ __bf16 hbf[BB * HID];  // 32 KB

  const int dir = blockIdx.x;
  const __bf16* Whh = dir ? WhhB : WhhF;
  const float* bhh = dir ? bhhB : bhhF;
  const float* Gi = dir ? GiB : GiF;
  float* hws = h_ws + dir * BB * HID;
  float* gh = gh_ws + dir * BB * 3 * HID;

  const int tid = threadIdx.x;
  const int wave = tid >> 5;
  const int lane = tid & 31;
  const int half = lane >> 4;
  const int mr = lane & 15;

  for (int i = tid; i < BB * HID; i += 1024) {
    gstoref(hws + i, 0.0f);
    hbf[i] = (__bf16)0.0f;
  }
  __threadfence_block();
  __syncthreads();

  for (int s = 0; s < SS; ++s) {
    const int xs = dir ? (SS - 1 - s) : s;

    // opaque copy: blocks LICM from hoisting weight loads across time steps
    const __bf16* Whh_s = Whh;
    asm volatile("" : "+s"(Whh_s));

    // ---- phase 1: gh = h @ Whh^T + bhh   (M=32, N=1536, K=512) ----
#pragma unroll 1
    for (int i = 0; i < 6; ++i) {
      const int tile = wave * 6 + i;
      const int tm = tile / 96;
      const int tn = tile - tm * 96;
      const __bf16* Abase = &hbf[(tm * 16 + mr) * HID];
      const __bf16* Wbase = Whh_s + (size_t)(tn * 16 + mr) * HID;
      f32x8 acc;
      const float binit = gloadf(bhh + tn * 16 + mr);
#pragma unroll
      for (int r = 0; r < 8; ++r) acc[r] = binit;

      bf16x16 av = load_op16_lds(Abase, half);
      bf16x16 bv = load_op16_g(Wbase, half);
      for (int kk = 32; kk < HID; kk += 32) {
        __builtin_prefetch(Wbase + kk + 96, 0, 1);
        bf16x16 avn = load_op16_lds(Abase + kk, half);
        bf16x16 bvn = load_op16_g(Wbase + kk, half);
        acc = wmma_bf16(av, bv, acc);
        av = avn;
        bv = bvn;
      }
      acc = wmma_bf16(av, bv, acc);

      const int col = tn * 16 + mr;
#pragma unroll
      for (int r = 0; r < 8; ++r)
        gstoref(gh + (tm * 16 + r + half * 8) * (3 * HID) + col, acc[r]);
    }
    __threadfence_block();
    __syncthreads();

    // ---- phase 2: fused GRU gate update ----
    for (int e = tid; e < BB * HID; e += 1024) {
      const int b = e >> 9;
      const int d = e & 511;
      const float* gi = Gi + (size_t)(xs * BB + b) * (3 * HID);
      const float* ghb = gh + b * (3 * HID);
      const float r = sigmoidf_(gloadf(gi + d) + gloadf(ghb + d));
      const float z =
          sigmoidf_(gloadf(gi + HID + d) + gloadf(ghb + HID + d));
      const float n =
          tanhf(gloadf(gi + 2 * HID + d) + r * gloadf(ghb + 2 * HID + d));
      const float hnew = (1.0f - z) * n + z * gloadf(hws + e);
      gstoref(hws + e, hnew);
      hbf[e] = (__bf16)hnew;
      const size_t o = ((size_t)b * SS + xs) * (2 * HID) + dir * HID + d;
      gstoref(enc_out + o, hnew);
      gstoreb(enc_out_bf + o, (__bf16)hnew);
      if (s == SS - 1)
        gstoreb(hcat + b * (2 * HID) + dir * HID + d, (__bf16)hnew);
    }
    __threadfence_block();
    __syncthreads();
  }
}

// ---------------------------------------------------------------------------
// Attention step: scores -> softmax -> context. One block per batch row.
// energy = tanh(attn_pre[b,s,:] + sterm[b,:]); score = energy . attn_v
// ---------------------------------------------------------------------------
__global__ __launch_bounds__(256) void attention_step_kernel(
    const float* __restrict__ attn_pre,  // [B*S, D], row b*S+s
    const float* __restrict__ sterm,     // [B, D]
    const float* __restrict__ attn_v,    // [D]
    const float* __restrict__ enc_out,   // [B, S, 2H]
    float* __restrict__ cbuf,            // [B, 2H]
    __bf16* __restrict__ zcat)           // [B, 2048]: cols 512..1535 <- c
{
  __shared__ float sc[SS];
  __shared__ float aw[SS];
  const int b = blockIdx.x;
  const int tid = threadIdx.x;
  const int wave = tid >> 5;
  const int lane = tid & 31;

  for (int si = wave; si < SS; si += 8) {
    float p = 0.0f;
    const float* ep = attn_pre + (size_t)(b * SS + si) * DDIM;
    const float* st = sterm + b * DDIM;
    for (int v = lane; v < DDIM; v += 32)
      p += tanhf(gloadf(ep + v) + gloadf(st + v)) * gloadf(attn_v + v);
#pragma unroll
    for (int off = 16; off > 0; off >>= 1) p += __shfl_down(p, off, 32);
    if (lane == 0) sc[si] = p;
  }
  __syncthreads();

  if (tid == 0) {
    float m = sc[0];
    for (int i = 1; i < SS; ++i) m = fmaxf(m, sc[i]);
    float sum = 0.0f;
    for (int i = 0; i < SS; ++i) {
      const float e = __expf(sc[i] - m);
      aw[i] = e;
      sum += e;
    }
    const float inv = 1.0f / sum;
    for (int i = 0; i < SS; ++i) aw[i] *= inv;
  }
  __syncthreads();

  for (int v = tid; v < 2 * HID; v += 256) {
    float acc = 0.0f;
    const float* eo = enc_out + (size_t)b * SS * (2 * HID) + v;
    for (int si = 0; si < SS; ++si) acc += aw[si] * gloadf(eo + si * (2 * HID));
    gstoref(cbuf + b * (2 * HID) + v, acc);
    gstoreb(zcat + b * 2048 + HID + v, (__bf16)acc);
  }
}

// ---------------------------------------------------------------------------
// Decoder GRU gate fusion + concat assembly for pred GEMM.
// gi = GiE[t] + GiC (bih in GiE); gh holds bhh. Updates s in place.
// zcat row = [s2 (512) | c (1024) | e (512)] bf16.
// ---------------------------------------------------------------------------
__global__ __launch_bounds__(512) void dec_gate_kernel(
    const float* __restrict__ GiE,   // [(T-1)*B, 3D]
    const float* __restrict__ GiC,   // [B, 3D]
    const float* __restrict__ Gh,    // [B, 3D]
    const __bf16* __restrict__ Xt,   // [(T-1)*B, E]
    float* __restrict__ sbuf,        // [B, D] in/out
    __bf16* __restrict__ s_bf,       // [B, D]
    __bf16* __restrict__ zcat, int t) {
  const int b = blockIdx.x;
  const int d = threadIdx.x;
  const float* gie = GiE + (size_t)(t * BB + b) * (3 * DDIM);
  const float* gic = GiC + b * (3 * DDIM);
  const float* gh = Gh + b * (3 * DDIM);
  const float gr = gloadf(gie + d) + gloadf(gic + d);
  const float gz = gloadf(gie + DDIM + d) + gloadf(gic + DDIM + d);
  const float gn = gloadf(gie + 2 * DDIM + d) + gloadf(gic + 2 * DDIM + d);
  const float r = sigmoidf_(gr + gloadf(gh + d));
  const float z = sigmoidf_(gz + gloadf(gh + DDIM + d));
  const float n = tanhf(gn + r * gloadf(gh + 2 * DDIM + d));
  const float hnew = (1.0f - z) * n + z * gloadf(sbuf + b * DDIM + d);
  gstoref(sbuf + b * DDIM + d, hnew);
  gstoreb(s_bf + b * DDIM + d, (__bf16)hnew);
  gstoreb(zcat + b * 2048 + d, (__bf16)hnew);
  gstoreb(zcat + b * 2048 + 1536 + d, gloadb(Xt + (size_t)(t * BB + b) * EMB + d));
}

// ---------------------------------------------------------------------------
// Host-side GEMM dispatch
// ---------------------------------------------------------------------------
static void launch_gemm(hipStream_t st, const __bf16* A, int lda,
                        const __bf16* W, int ldw, const float* bias, float* C,
                        int ldc, __bf16* Cbf, int ldcbf, int M, int N, int K,
                        int act) {
  const bool hb = bias != nullptr;
  const bool wb = Cbf != nullptr;
  if (M == 32) {
    const dim3 g((N >> 4) / 8), blk(256);
    const size_t shmem = (size_t)32 * K * sizeof(__bf16);
    if (act == 0) {
      if (hb && wb)
        gemm_wmma_m32_kernel<0, true, true><<<g, blk, shmem, st>>>(A, lda, W, ldw, bias, C, ldc, Cbf, ldcbf, N, K);
      else if (hb)
        gemm_wmma_m32_kernel<0, true, false><<<g, blk, shmem, st>>>(A, lda, W, ldw, bias, C, ldc, Cbf, ldcbf, N, K);
      else if (wb)
        gemm_wmma_m32_kernel<0, false, true><<<g, blk, shmem, st>>>(A, lda, W, ldw, bias, C, ldc, Cbf, ldcbf, N, K);
      else
        gemm_wmma_m32_kernel<0, false, false><<<g, blk, shmem, st>>>(A, lda, W, ldw, bias, C, ldc, Cbf, ldcbf, N, K);
    } else {
      if (hb && wb)
        gemm_wmma_m32_kernel<1, true, true><<<g, blk, shmem, st>>>(A, lda, W, ldw, bias, C, ldc, Cbf, ldcbf, N, K);
      else if (hb)
        gemm_wmma_m32_kernel<1, true, false><<<g, blk, shmem, st>>>(A, lda, W, ldw, bias, C, ldc, Cbf, ldcbf, N, K);
      else if (wb)
        gemm_wmma_m32_kernel<1, false, true><<<g, blk, shmem, st>>>(A, lda, W, ldw, bias, C, ldc, Cbf, ldcbf, N, K);
      else
        gemm_wmma_m32_kernel<1, false, false><<<g, blk, shmem, st>>>(A, lda, W, ldw, bias, C, ldc, Cbf, ldcbf, N, K);
    }
    return;
  }
  const int tiles = (M >> 4) * (N >> 4);
  const dim3 g((tiles + 7) / 8), blk(256);
  const size_t shmem = (size_t)16 * K * sizeof(__bf16);
  if (act == 0) {
    if (hb)
      gemm_wmma_kernel<0, true, false><<<g, blk, shmem, st>>>(A, lda, W, ldw, bias, C, ldc, Cbf, ldcbf, M, N, K);
    else
      gemm_wmma_kernel<0, false, false><<<g, blk, shmem, st>>>(A, lda, W, ldw, bias, C, ldc, Cbf, ldcbf, M, N, K);
  } else {
    if (hb)
      gemm_wmma_kernel<1, true, false><<<g, blk, shmem, st>>>(A, lda, W, ldw, bias, C, ldc, Cbf, ldcbf, M, N, K);
    else
      gemm_wmma_kernel<1, false, false><<<g, blk, shmem, st>>>(A, lda, W, ldw, bias, C, ldc, Cbf, ldcbf, M, N, K);
  }
}

static void launch_f2bf(hipStream_t st, const float* s, __bf16* d, int n) {
  f2bf_kernel<<<(n + 255) / 256, 256, 0, st>>>(s, d, n);
}

// ---------------------------------------------------------------------------
// kernel_launch
// ---------------------------------------------------------------------------
extern "C" void kernel_launch(void* const* d_in, const int* in_sizes, int n_in,
                              void* d_out, int out_size, void* d_ws,
                              size_t ws_size, hipStream_t stream) {
  (void)in_sizes; (void)n_in; (void)out_size; (void)ws_size;

  const int*   src      = (const int*)d_in[0];
  const int*   trg      = (const int*)d_in[1];
  const float* emb      = (const float*)d_in[2];
  const float* encf_Wih = (const float*)d_in[3];
  const float* encf_Whh = (const float*)d_in[4];
  const float* encf_bih = (const float*)d_in[5];
  const float* encf_bhh = (const float*)d_in[6];
  const float* encb_Wih = (const float*)d_in[7];
  const float* encb_Whh = (const float*)d_in[8];
  const float* encb_bih = (const float*)d_in[9];
  const float* encb_bhh = (const float*)d_in[10];
  const float* fc_W     = (const float*)d_in[11];
  const float* fc_b     = (const float*)d_in[12];
  const float* attn_W   = (const float*)d_in[13];
  const float* attn_v   = (const float*)d_in[14];
  const float* dec_Wih  = (const float*)d_in[15];
  const float* dec_Whh  = (const float*)d_in[16];
  const float* dec_bih  = (const float*)d_in[17];
  const float* dec_bhh  = (const float*)d_in[18];
  const float* out_W    = (const float*)d_in[19];
  const float* out_b    = (const float*)d_in[20];
  float* out = (float*)d_out;

  // ---- workspace bump allocator (256B aligned) ----
  char* ws = (char*)d_ws;
  size_t off = 0;
  auto alloc = [&](size_t bytes) -> void* {
    off = (off + 255) & ~(size_t)255;
    void* p = ws + off;
    off += bytes;
    return p;
  };
  auto abf = [&](size_t n) { return (__bf16*)alloc(n * sizeof(__bf16)); };
  auto af32 = [&](size_t n) { return (float*)alloc(n * sizeof(float)); };

  __bf16* encfWih_bf = abf(3 * HID * EMB);
  __bf16* encfWhh_bf = abf(3 * HID * HID);
  __bf16* encbWih_bf = abf(3 * HID * EMB);
  __bf16* encbWhh_bf = abf(3 * HID * HID);
  __bf16* fcW_bf     = abf(DDIM * 2 * HID);
  __bf16* attnW_bf   = abf(DDIM * (DDIM + 2 * HID));
  __bf16* decWih_bf  = abf(3 * DDIM * (EMB + 2 * HID));
  __bf16* decWhh_bf  = abf(3 * DDIM * DDIM);
  __bf16* outW_bf    = abf((size_t)VOC * 2048);
  __bf16* Xs         = abf((size_t)SS * BB * EMB);           // [s*B+b, E]
  __bf16* Xt         = abf((size_t)(TT - 1) * BB * EMB);     // [t*B+b, E]
  __bf16* enc_out_bf = abf((size_t)BB * SS * 2 * HID);
  __bf16* hcat       = abf(BB * 2 * HID);
  __bf16* s_bf       = abf(BB * DDIM);
  __bf16* zcat       = abf(BB * 2048);

  float* GiF      = af32((size_t)SS * BB * 3 * HID);
  float* GiB      = af32((size_t)SS * BB * 3 * HID);
  float* enc_out  = af32((size_t)BB * SS * 2 * HID);
  float* attn_pre = af32((size_t)BB * SS * DDIM);
  float* GiE      = af32((size_t)(TT - 1) * BB * 3 * DDIM);
  float* gh_ws    = af32(2 * BB * 3 * HID);
  float* h_ws     = af32(2 * BB * HID);
  float* sbuf     = af32(BB * DDIM);
  float* sterm    = af32(BB * DDIM);
  float* cbuf     = af32(BB * 2 * HID);
  float* GiC      = af32(BB * 3 * DDIM);
  float* Gh       = af32(BB * 3 * DDIM);

  // outputs[0] is all zeros in the reference
  hipMemsetAsync(out, 0, (size_t)BB * VOC * sizeof(float), stream);

  // ---- weight conversions to bf16 ----
  launch_f2bf(stream, encf_Wih, encfWih_bf, 3 * HID * EMB);
  launch_f2bf(stream, encf_Whh, encfWhh_bf, 3 * HID * HID);
  launch_f2bf(stream, encb_Wih, encbWih_bf, 3 * HID * EMB);
  launch_f2bf(stream, encb_Whh, encbWhh_bf, 3 * HID * HID);
  launch_f2bf(stream, fc_W, fcW_bf, DDIM * 2 * HID);
  launch_f2bf(stream, attn_W, attnW_bf, DDIM * (DDIM + 2 * HID));
  launch_f2bf(stream, dec_Wih, decWih_bf, 3 * DDIM * (EMB + 2 * HID));
  launch_f2bf(stream, dec_Whh, decWhh_bf, 3 * DDIM * DDIM);
  launch_f2bf(stream, out_W, outW_bf, (int)((size_t)VOC * 2048));

  // ---- embedding gathers ----
  gather_embed_kernel<<<SS * BB, 256, 0, stream>>>(src, emb, Xs, 0);
  gather_embed_kernel<<<(TT - 1) * BB, 256, 0, stream>>>(trg, emb, Xt, 1);

  // ---- encoder input GEMMs: Gi = X @ Wih^T + bih  (M=2048,N=1536,K=512) ----
  launch_gemm(stream, Xs, EMB, encfWih_bf, EMB, encf_bih, GiF, 3 * HID,
              nullptr, 0, SS * BB, 3 * HID, EMB, 0);
  launch_gemm(stream, Xs, EMB, encbWih_bf, EMB, encb_bih, GiB, 3 * HID,
              nullptr, 0, SS * BB, 3 * HID, EMB, 0);

  // ---- bidirectional GRU scan (2 workgroups, one per direction) ----
  encoder_scan_kernel<<<2, 1024, 0, stream>>>(
      encfWhh_bf, encbWhh_bf, encf_bhh, encb_bhh, GiF, GiB, enc_out,
      enc_out_bf, hcat, h_ws, gh_ws);

  // ---- step-invariant attention term: attn_pre = enc_out @ Wa_e^T ----
  // Wa_e = attn_W[:, D:D+2H]  (row stride 1536, offset 512)
  launch_gemm(stream, enc_out_bf, 2 * HID, attnW_bf + DDIM, DDIM + 2 * HID,
              nullptr, attn_pre, DDIM, nullptr, 0, BB * SS, DDIM, 2 * HID, 0);

  // ---- step-invariant decoder-input term: GiE = Xt @ Wih[:, :E]^T + bih ----
  launch_gemm(stream, Xt, EMB, decWih_bf, EMB + 2 * HID, dec_bih, GiE,
              3 * DDIM, nullptr, 0, (TT - 1) * BB, 3 * DDIM, EMB, 0);

  // ---- s0 = tanh([hf,hb] @ fc_W^T + fc_b) ----
  launch_gemm(stream, hcat, 2 * HID, fcW_bf, 2 * HID, fc_b, sbuf, DDIM, s_bf,
              DDIM, BB, DDIM, 2 * HID, 1);

  // ---- decoder loop ----
  for (int t = 0; t < TT - 1; ++t) {
    // sterm = s @ Wa_s^T   (Wa_s = attn_W[:, :D])
    launch_gemm(stream, s_bf, DDIM, attnW_bf, DDIM + 2 * HID, nullptr, sterm,
                DDIM, nullptr, 0, BB, DDIM, DDIM, 0);
    // attention: scores, softmax, context c  (writes zcat[:,512:1536])
    attention_step_kernel<<<BB, 256, 0, stream>>>(attn_pre, sterm, attn_v,
                                                  enc_out, cbuf, zcat);
    // GiC = c @ Wih[:, E:]^T   (A = zcat + 512, lda=2048, K=1024)
    launch_gemm(stream, zcat + HID, 2048, decWih_bf + EMB, EMB + 2 * HID,
                nullptr, GiC, 3 * DDIM, nullptr, 0, BB, 3 * DDIM, 2 * HID, 0);
    // Gh = s @ dec_Whh^T + bhh
    launch_gemm(stream, s_bf, DDIM, decWhh_bf, DDIM, dec_bhh, Gh, 3 * DDIM,
                nullptr, 0, BB, 3 * DDIM, DDIM, 0);
    // gate fusion: s <- GRU(s); zcat row = [s2 | c | e]
    dec_gate_kernel<<<BB, DDIM, 0, stream>>>(GiE, GiC, Gh, Xt, sbuf, s_bf,
                                             zcat, t);
    // pred[t+1] = zcat @ out_W^T + out_b    (M=32, N=32000, K=2048)
    launch_gemm(stream, zcat, 2048, outW_bf, 2048, out_b,
                out + (size_t)(t + 1) * BB * VOC, VOC, nullptr, 0, BB, VOC,
                2048, 0);
  }
}
